// Conv1dGroup_34909494182542
// MI455X (gfx1250) — matrix operand order repaired and checked by reading, not compiled, not run on hardware
//
#include <hip/hip_runtime.h>

typedef __attribute__((ext_vector_type(16))) __bf16 v16bf;
typedef __attribute__((ext_vector_type(8)))  float  v8f;
typedef __attribute__((ext_vector_type(8)))  unsigned int v8u;

union ABits { v8u u; v16bf b; };

#define NG 129      // groups
#define NTT 512     // T
#define NB 4        // batch
#define NF 96       // in features per group
#define NO 96      // out features per group
#define NK 3        // kernel taps
#define RD 288      // reduction dim = NF*NK
#define OB 48       // O columns per block (2 blocks cover O)
#define WS_DW 145   // weight LDS per-o stride in dwords (odd => bank-friendly)
#define MROWS 256   // output rows per block
#define XROWS 258   // staged x rows = MROWS + 2 halo
#define XS_DW 49    // x LDS row stride in dwords (odd => bank-friendly)

__global__ __launch_bounds__(256) void conv1d_group_wmma(
    const float* __restrict__ x, const float* __restrict__ w,
    const float* __restrict__ bias, float* __restrict__ out)
{
  __shared__ unsigned int sbh[OB * WS_DW];     // weight hi bf16 pairs, [o][r=k*NF+f]
  __shared__ unsigned int sbl[OB * WS_DW];     // weight lo (residual) bf16 pairs
  __shared__ unsigned int sxh[XROWS * XS_DW];  // x hi bf16 pairs, [t'][f]
  __shared__ unsigned int sxl[XROWS * XS_DW];  // x lo (residual) bf16 pairs

  const int g     = blockIdx.x;
  const int oBase = blockIdx.z * OB;
  const int tid   = threadIdx.x;
  const int lane  = tid & 31;
  const int wave  = tid >> 5;
  const int mbase = blockIdx.y * MROWS;     // first output row of this block
  const int tb    = mbase & (NTT - 1);      // t of first row (whole block shares b)
  const int btb   = mbase - tb;             // b * T

  // -------- stage x tile (rows tb-1 .. tb+256) as bf16 hi/lo, once per element
  for (int idx = tid; idx < XROWS * (NF / 4); idx += 256) {
    const int row_l = idx / (NF / 4);
    const int fq    = idx - row_l * (NF / 4);     // float4 index within row
    const int tp    = tb - 1 + row_l;
    float4 q = {0.0f, 0.0f, 0.0f, 0.0f};          // zero padding rows
    if (tp >= 0 && tp < NTT)
      q = *(const float4*)(x + (((size_t)(btb + tp) * NG + g) * NF + 4 * fq));
    unsigned int hA = __float_as_uint(q.x) & 0xFFFF0000u;
    unsigned int hB = __float_as_uint(q.y) & 0xFFFF0000u;
    unsigned int hC = __float_as_uint(q.z) & 0xFFFF0000u;
    unsigned int hD = __float_as_uint(q.w) & 0xFFFF0000u;
    float rA = q.x - __uint_as_float(hA);
    float rB = q.y - __uint_as_float(hB);
    float rC = q.z - __uint_as_float(hC);
    float rD = q.w - __uint_as_float(hD);
    const int dbase = row_l * XS_DW + 2 * fq;
    sxh[dbase]     = (hA >> 16) | hB;
    sxh[dbase + 1] = (hC >> 16) | hD;
    sxl[dbase]     = (__float_as_uint(rA) >> 16) | (__float_as_uint(rB) & 0xFFFF0000u);
    sxl[dbase + 1] = (__float_as_uint(rC) >> 16) | (__float_as_uint(rD) & 0xFFFF0000u);
  }

  // -------- stage group weights as bf16 hi/lo, layout [o][r=k*NF+f]
  {
    unsigned short* sh = (unsigned short*)sbh;
    unsigned short* sl = (unsigned short*)sbl;
    const float* wg = w + ((size_t)g * NO + oBase) * RD;
    for (int wi = tid; wi < OB * RD; wi += 256) {    // coalesced global reads
      int o_l = wi / RD;
      int rem = wi - o_l * RD;                       // memory order: f*NK + k
      int f   = rem / NK;
      int k   = rem - f * NK;
      int r   = k * NF + f;                          // WMMA reduction order
      float v = wg[wi];
      unsigned int hb = __float_as_uint(v) & 0xFFFF0000u;
      float resid = v - __uint_as_float(hb);
      sh[o_l * (2 * WS_DW) + r] = (unsigned short)(hb >> 16);
      sl[o_l * (2 * WS_DW) + r] = (unsigned short)(__float_as_uint(resid) >> 16);
    }
  }
  __syncthreads();

  const int o_l  = lane & 15;
  const int mrow = mbase + wave * 32;

  // init accumulators with broadcast bias (all rows in a lane share column o)
  v8f acc[2][3];
  #pragma unroll
  for (int nt = 0; nt < 3; ++nt) {
    const float bv = bias[g * NO + oBase + nt * 16 + o_l];
    #pragma unroll
    for (int u = 0; u < 2; ++u)
      #pragma unroll
      for (int j = 0; j < 8; ++j)
        acc[u][nt][j] = bv;
  }

  #pragma unroll
  for (int s = 0; s < 9; ++s) {          // K=288 in 9 steps of 32; each step = 32 features of one tap
    const int k  = s / 3;
    const int f0 = (s - k * 3) * 32;
    const int adw = (f0 >> 1) + ((lane < 16) ? 0 : 4);   // A K-runs {0..7}/{16..23} (+8 hi lanes)

    // A fragments (two 16x32 bf16 tiles, hi/lo) from LDS
    ABits ah[2], al[2];
    #pragma unroll
    for (int u = 0; u < 2; ++u) {
      const int row_l = wave * 32 + u * 16 + (lane & 15) + k;  // (t + k - 1) - (tb - 1)
      const unsigned int* pxh = &sxh[row_l * XS_DW + adw];
      const unsigned int* pxl = &sxl[row_l * XS_DW + adw];
      #pragma unroll
      for (int i = 0; i < 4; ++i) {
        ah[u].u[i] = pxh[i];  ah[u].u[4 + i] = pxh[8 + i];
        al[u].u[i] = pxl[i];  al[u].u[4 + i] = pxl[8 + i];
      }
    }

    // B fragments for all three N-tiles (hi/lo) from LDS
    const int bbase = 16 * s + ((lane < 16) ? 0 : 4);  // K-runs {0..7}/{16..23} (+8 hi lanes)
    ABits bh[3], bl[3];
    #pragma unroll
    for (int nt = 0; nt < 3; ++nt) {
      const unsigned int* ph = &sbh[(nt * 16 + o_l) * WS_DW + bbase];
      const unsigned int* pl = &sbl[(nt * 16 + o_l) * WS_DW + bbase];
      #pragma unroll
      for (int i = 0; i < 4; ++i) { bh[nt].u[i] = ph[i]; bh[nt].u[4 + i] = ph[8 + i]; bl[nt].u[i] = pl[i]; bl[nt].u[4 + i] = pl[8 + i]; }
    }

    // 18 WMMAs grouped by product type: 6 independent chains between any
    // dependent pair => no hazard NOPs.  hi*hi + hi*lo + lo*hi (lo*lo dropped)
    #pragma unroll
    for (int nt = 0; nt < 3; ++nt)
      #pragma unroll
      for (int u = 0; u < 2; ++u)
        acc[u][nt] = __builtin_amdgcn_wmma_f32_16x16x32_bf16(false, ah[u].b, false, bh[nt].b, (short)0, acc[u][nt], false, false);
    #pragma unroll
    for (int nt = 0; nt < 3; ++nt)
      #pragma unroll
      for (int u = 0; u < 2; ++u)
        acc[u][nt] = __builtin_amdgcn_wmma_f32_16x16x32_bf16(false, ah[u].b, false, bl[nt].b, (short)0, acc[u][nt], false, false);
    #pragma unroll
    for (int nt = 0; nt < 3; ++nt)
      #pragma unroll
      for (int u = 0; u < 2; ++u)
        acc[u][nt] = __builtin_amdgcn_wmma_f32_16x16x32_bf16(false, al[u].b, false, bh[nt].b, (short)0, acc[u][nt], false, false);
  }

  // -------- store (C/D: VGPR j -> M = j + 8*(lane>=16); lane%16 -> N); bias already in acc
  #pragma unroll
  for (int nt = 0; nt < 3; ++nt) {
    const int o = oBase + nt * 16 + o_l;
    #pragma unroll
    for (int u = 0; u < 2; ++u) {
      #pragma unroll
      for (int j = 0; j < 8; ++j) {
        const int row = mrow + u * 16 + j + ((lane >> 4) << 3);
        out[((size_t)row * NG + g) * NO + o] = acc[u][nt][j];
      }
    }
  }
}

extern "C" void kernel_launch(void* const* d_in, const int* in_sizes, int n_in,
                              void* d_out, int out_size, void* d_ws, size_t ws_size,
                              hipStream_t stream) {
  (void)in_sizes; (void)n_in; (void)out_size; (void)d_ws; (void)ws_size;
  const float* x  = (const float*)d_in[0];
  const float* w  = (const float*)d_in[1];
  const float* bs = (const float*)d_in[2];
  float* out = (float*)d_out;
  dim3 grid(NG, (NB * NTT) / MROWS, NO / OB);   // (129, 8, 2)
  conv1d_group_wmma<<<grid, dim3(256), 0, stream>>>(x, w, bs, out);
}
